// VectorQuantizer_43559558316121
// MI455X (gfx1250) — matrix operand order repaired and checked by reading, not compiled, not run on hardware
//
#include <hip/hip_runtime.h>

typedef _Float16 v16h __attribute__((ext_vector_type(16)));
typedef float    v8f  __attribute__((ext_vector_type(8)));
typedef float    v4f  __attribute__((ext_vector_type(4)));

#define K_CODES 512
#define D_DIM   64
#define N_ROWS  (64*64*64)      // 262144 rows of length 64
#define STRIPS  (N_ROWS/16)     // 16384 strips of 16 rows
#define BLOCKS  256
#define THREADS 256
#define WAVES_PER_BLOCK (THREADS/32)
#define TOTAL_WAVES (BLOCKS*WAVES_PER_BLOCK)

// ---- prep: 0.5*||e_k||^2 -> ws[0..511], zero the loss slot ----
__global__ __launch_bounds__(256)
void vq_prep_kernel(const float* __restrict__ E, float* __restrict__ wsen,
                    float* __restrict__ lossp)
{
    const int k = blockIdx.x * 256 + threadIdx.x;   // 512 codes total
    float s = 0.f;
    #pragma unroll 8
    for (int d = 0; d < D_DIM; ++d) {
        float e = E[d * K_CODES + k];
        s += e * e;
    }
    wsen[k] = 0.5f * s;
    if (k == 0) *lossp = 0.0f;
}

__global__ __launch_bounds__(THREADS)
void vq_wmma_kernel(const float* __restrict__ x,
                    const float* __restrict__ E,      // [64][512] f32
                    const float* __restrict__ wsen,   // [512] 0.5*||e_k||^2
                    float* __restrict__ out,          // [N][64] f32
                    float* __restrict__ lossp)
{
    // Exactly 64 KB of LDS: 64 pre-swizzled f16 B-operand tiles (1 KB each).
    __shared__ unsigned int sB[16384];

    const int tid    = threadIdx.x;
    const int lane   = tid & 31;
    const int waveId = tid >> 5;
    const int hi     = lane >> 4;   // 0/1 half of the wave
    const int ln     = lane & 15;
    const int koff   = hi * 8;      // A-operand K offset for this lane half

    // ---------------- codebook -> f16 B operands in LDS ----------------
    // B operand layout (V_WMMA_F32_16X16X32_F16, wave32): lane: N = lane%16,
    // elements in table K order (K = 8*(lane/16) + s, s<8; 16 + 8*(lane/16) + s-8, s>=8).
    // op = 2*t + kh covers codes [16t,16t+16) x D [32*kh, 32*kh+32).
    for (int p = tid; p < 64 * 32; p += THREADS) {
        const int op    = p >> 5;
        const int plane = p & 31;        // lane slot: N = plane%16, K half = plane/16
        const int t     = op >> 1;
        const int kh    = op & 1;
        const int d     = 32 * kh + 8 * (plane >> 4);  // reduction (D) base index
        const float* src = E + d * K_CODES + t * 16 + (plane & 15); // code N
        unsigned int* dst = sB + op * 256 + plane * 8;
        #pragma unroll
        for (int j = 0; j < 8; ++j) {
            union { _Float16 h[2]; unsigned int w; } u;
            const int kk = 2 * j + (j >= 4 ? 8 : 0);
            u.h[0] = (_Float16)src[kk * K_CODES];
            u.h[1] = (_Float16)src[(kk + 1) * K_CODES];
            dst[j] = u.w;
        }
    }
    __syncthreads();

    // ---------------- per-wave strips of 16 rows ----------------
    const int gwave = blockIdx.x * WAVES_PER_BLOCK + waveId;
    float lsum = 0.0f;

    for (int strip = gwave; strip < STRIPS; strip += TOTAL_WAVES) {
        if (strip + TOTAL_WAVES < STRIPS)
            __builtin_prefetch(x + (size_t)(strip + TOTAL_WAVES) * 16 * 64, 0, 1);

        // ---- Load A operands (NEGATED x rows, f32 -> f16, ISA A 16x32 layout)
        // lane: M = ln, VGPR0-3 = K in [koff,koff+8), VGPR4-7 = K in [16+koff, ...)
        // A holds -x so that  C=0.5*||e||^2  accumulates to  0.5*||e||^2 - x.e,
        // which orders identically to the squared distance.
        const int arow = strip * 16 + ln;
        const float* xr = x + (size_t)arow * 64;
        v4f xc[8];
        #pragma unroll
        for (int c = 0; c < 4; ++c) {
            const float* pc = xr + c * 16 + koff;
            xc[2 * c]     = *(const v4f*)(pc);
            xc[2 * c + 1] = *(const v4f*)(pc + 4);
        }
        v16h a0, a1;
        #pragma unroll
        for (int e = 0; e < 4; ++e) {
            a0[e]      = (_Float16)(-xc[0][e]);
            a0[4 + e]  = (_Float16)(-xc[1][e]);
            a0[8 + e]  = (_Float16)(-xc[2][e]);
            a0[12 + e] = (_Float16)(-xc[3][e]);
            a1[e]      = (_Float16)(-xc[4][e]);
            a1[4 + e]  = (_Float16)(-xc[5][e]);
            a1[8 + e]  = (_Float16)(-xc[6][e]);
            a1[12 + e] = (_Float16)(-xc[7][e]);
        }

        // ---- GEMM + running argmin over 512 codes (32 tiles x 2 WMMAs)
        float minv[8];
        int   mini[8];
        #pragma unroll
        for (int v = 0; v < 8; ++v) { minv[v] = 3.4e38f; mini[v] = 0; }

        #pragma unroll 4
        for (int t = 0; t < 32; ++t) {
            const unsigned int* bbase = sB + (t * 2) * 256 + lane * 8;
            v16h b0 = *(const v16h*)(bbase);
            v16h b1 = *(const v16h*)(bbase + 256);
            const float ek = wsen[t * 16 + ln];      // L0-resident after 1st pass
            v8f acc = {ek, ek, ek, ek, ek, ek, ek, ek};
            acc = __builtin_amdgcn_wmma_f32_16x16x32_f16(
                      false, a0, false, b0, (short)0, acc, false, false);
            acc = __builtin_amdgcn_wmma_f32_16x16x32_f16(
                      false, a1, false, b1, (short)0, acc, false, false);
            const int kcol = t * 16 + ln;            // C layout: lane = column N
            #pragma unroll
            for (int v = 0; v < 8; ++v) {            // VGPR v = row M = 8*hi + v
                if (acc[v] < minv[v]) { minv[v] = acc[v]; mini[v] = kcol; }
            }
        }

        // ---- min-reduce across the 16 lanes sharing each row group
        #pragma unroll
        for (int v = 0; v < 8; ++v) {
            float mv = minv[v];
            int   mi = mini[v];
            #pragma unroll
            for (int off = 1; off < 16; off <<= 1) {
                float ov = __shfl_xor(mv, off, 32);
                int   oi = __shfl_xor(mi, off, 32);
                if (ov < mv || (ov == mv && oi < mi)) { mv = ov; mi = oi; }
            }
            minv[v] = mv;
            mini[v] = mi;   // replicated across the 16 lanes of this half
        }

        // ---- gather codebook rows (exact f32), write output, accumulate loss
        // lane handles d = 4*ln .. 4*ln+3 of rows m = 8*hi + v
        const int dp = 4 * ln;
        #pragma unroll
        for (int v = 0; v < 8; ++v) {
            const int m    = 8 * hi + v;
            const int idx  = mini[v];
            const size_t ob = (size_t)(strip * 16 + m) * 64 + dp;
            v4f q;
            #pragma unroll
            for (int c = 0; c < 4; ++c) q[c] = E[(dp + c) * K_CODES + idx];
            v4f xv = *(const v4f*)(x + ob);
            v4f df = q - xv;
            lsum += df[0] * df[0] + df[1] * df[1] + df[2] * df[2] + df[3] * df[3];
            *(v4f*)(out + ob) = q;
        }
    }

    // ---- loss: wave reduce, one agent-scope f32 atomic per wave
    #pragma unroll
    for (int off = 1; off < 32; off <<= 1) lsum += __shfl_xor(lsum, off, 32);
    if (lane == 0) {
        // loss = (BETA + 1) * mean((q - x)^2) = 1.25 * sum / (N*D)
        __hip_atomic_fetch_add(lossp, lsum * (1.25f / 16777216.0f),
                               __ATOMIC_RELAXED, __HIP_MEMORY_SCOPE_AGENT);
    }
}

extern "C" void kernel_launch(void* const* d_in, const int* in_sizes, int n_in,
                              void* d_out, int out_size, void* d_ws, size_t ws_size,
                              hipStream_t stream) {
    const float* x = (const float*)d_in[0];   // [64,64,64,64] f32
    const float* E = (const float*)d_in[1];   // [64,512] f32
    float* out   = (float*)d_out;             // [N*64] quantized_st ++ [1] loss
    float* lossp = out + (out_size - 1);
    float* wsen  = (float*)d_ws;              // 512 floats of scratch

    vq_prep_kernel<<<2, 256, 0, stream>>>(E, wsen, lossp);
    vq_wmma_kernel<<<BLOCKS, THREADS, 0, stream>>>(x, E, wsen, out, lossp);
}
